// MultiheadAttention_84902913507815
// MI455X (gfx1250) — matrix operand order repaired and checked by reading, not compiled, not run on hardware
//
#include <hip/hip_runtime.h>
#include <cstddef>

// ---- CDNA5 WMMA types ----
typedef __attribute__((ext_vector_type(16))) __bf16 v16bf;
typedef __attribute__((ext_vector_type(8)))  float  v8f;

struct alignas(16) U128 { unsigned int x, y, z, w; };

union Frag {
    v16bf v;
    U128 q[2];
    unsigned short s[16];
};

// raw 16-float A-tile slice (two contiguous 8-float runs), held across pipeline stages
struct AF32 { U128 r[4]; };
union AF32View { AF32 a; float f[16]; };

#define D_DIM 768
#define T_DIM 1024
#define B_DIM 8
#define H_NUM 12
#define HD    64
#define NSUP  (D_DIM / 64)        // 12 N-super-tiles (64 wide = one head)
#define NHEAD (B_DIM * H_NUM)     // 96

__device__ __forceinline__ unsigned short f2bf(float f) {
    unsigned int u = __builtin_bit_cast(unsigned int, f);
    u += 0x7FFFu + ((u >> 16) & 1u);          // round-to-nearest-even
    return (unsigned short)(u >> 16);
}

// B-fragment: two 8-element K runs per lane (lanes 0-15: K=0..7,16..23; 16-31: K=8..15,24..31)
__device__ __forceinline__ v16bf load_b_frag(const unsigned short* p) {
    Frag f;
    f.q[0] = *(const U128*)p;
    f.q[1] = *(const U128*)(p + 16);
    return f.v;
}
// A-fragment: two contiguous 8-element K runs per lane
__device__ __forceinline__ v16bf load_a_frag(const unsigned short* p0, const unsigned short* p1) {
    Frag f;
    f.q[0] = *(const U128*)p0;
    f.q[1] = *(const U128*)p1;
    return f.v;
}
// raw f32 A slice load (no conversion yet)
__device__ __forceinline__ AF32 load_a_raw(const float* p0, const float* p1) {
    AF32 a;
    a.r[0] = *(const U128*)p0;
    a.r[1] = *(const U128*)(p0 + 4);
    a.r[2] = *(const U128*)p1;
    a.r[3] = *(const U128*)(p1 + 4);
    return a;
}
// convert raw f32 slice (already in regs) to bf16 A-fragment
__device__ __forceinline__ v16bf cvt_a_raw(const AF32& a) {
    AF32View v; v.a = a;
    Frag f;
    #pragma unroll
    for (int i = 0; i < 16; ++i) f.s[i] = f2bf(v.f[i]);
    return f.v;
}

__device__ __forceinline__ v8f wmma_bf16(v16bf a, v16bf b, v8f c) {
    return __builtin_amdgcn_wmma_f32_16x16x32_bf16(false, a, false, b, (short)0, c, false, false);
}

// ---------------- f32 -> bf16 convert ----------------
__global__ void __launch_bounds__(256)
cvt_f32_bf16_kernel(const float* __restrict__ src, unsigned short* __restrict__ dst, int n) {
    int i = blockIdx.x * blockDim.x + threadIdx.x;
    if (i < n) dst[i] = f2bf(src[i]);
}

// ---------------- fused Q/K/V projection ----------------
// grid: (768, 1, 3).  blockIdx.z: 0=q, 1=k, 2=v.
// One wave per 16x64 tile, software-pipelined: next A(raw f32) + 4 B frags load
// while current 4 WMMAs issue.  Outputs: Qh/Kh (head, T, hd); Vt (head, hd, T).
__global__ void __launch_bounds__(256)
qkv_proj_kernel(const float* __restrict__ Xq, const float* __restrict__ Xk,
                const float* __restrict__ Xv,
                const unsigned short* __restrict__ Wbf,   // 7 x 768x768 bf16: wq,wk,wv,wqm,wkm,wvm,wo
                const float* __restrict__ bq, const float* __restrict__ bk,
                const float* __restrict__ bv,
                const float* __restrict__ bqm, const float* __restrict__ bkm,
                const float* __restrict__ bvm,
                unsigned short* __restrict__ Qh, unsigned short* __restrict__ Kh,
                unsigned short* __restrict__ Vt,
                const int* __restrict__ ls_len_p)
{
    const int lane = threadIdx.x & 31;
    const int warp = threadIdx.x >> 5;
    const int wid  = blockIdx.x * 8 + warp;     // 0..6143 = 512 Mtiles * 12 Nsuper
    const int m0   = (wid / NSUP) * 16;
    const int nsup = wid % NSUP;                // == head index within D
    const int n0   = nsup * 64;
    const int z    = blockIdx.z;
    const int hi   = lane >> 4;
    const int ln   = lane & 15;
    const int lsl  = *ls_len_p;

    const float* X = (z == 0) ? Xq : (z == 1) ? Xk : Xv;
    const bool main_part = ((m0 >> 3) < lsl);   // row m -> t = m / B
    const unsigned short* W =
        Wbf + (size_t)(main_part ? z : z + 3) * (size_t)(D_DIM * D_DIM);
    const float* bias = main_part ? ((z == 0) ? bq : (z == 1) ? bk : bv)
                                  : ((z == 0) ? bqm : (z == 1) ? bkm : bvm);

    const float* xrow = X + (size_t)(m0 + ln) * D_DIM;
    const unsigned short* wr0 = W + (size_t)(n0 + ln)      * D_DIM;
    const unsigned short* wr1 = W + (size_t)(n0 + 16 + ln) * D_DIM;
    const unsigned short* wr2 = W + (size_t)(n0 + 32 + ln) * D_DIM;
    const unsigned short* wr3 = W + (size_t)(n0 + 48 + ln) * D_DIM;

    const v8f zero8 = {0.f, 0.f, 0.f, 0.f, 0.f, 0.f, 0.f, 0.f};
    v8f acc[4] = {zero8, zero8, zero8, zero8};
    const int ka = hi ? 8 : 0;
    const int kb = hi ? 8 : 0;

    // ---- software pipeline: stage 0 loads ----
    AF32  araw = load_a_raw(xrow + ka, xrow + ka + 16);
    v16bf b0 = load_b_frag(wr0 + kb);
    v16bf b1 = load_b_frag(wr1 + kb);
    v16bf b2 = load_b_frag(wr2 + kb);
    v16bf b3 = load_b_frag(wr3 + kb);

    #pragma unroll 2
    for (int kk = 0; kk < D_DIM - 32; kk += 32) {
        const int kn = kk + 32;
        AF32  an  = load_a_raw(xrow + kn + ka, xrow + kn + ka + 16);
        v16bf nb0 = load_b_frag(wr0 + kn + kb);
        v16bf nb1 = load_b_frag(wr1 + kn + kb);
        v16bf nb2 = load_b_frag(wr2 + kn + kb);
        v16bf nb3 = load_b_frag(wr3 + kn + kb);

        const v16bf a = cvt_a_raw(araw);       // data already in regs: no mem wait
        acc[0] = wmma_bf16(a, b0, acc[0]);
        acc[1] = wmma_bf16(a, b1, acc[1]);
        acc[2] = wmma_bf16(a, b2, acc[2]);
        acc[3] = wmma_bf16(a, b3, acc[3]);

        araw = an; b0 = nb0; b1 = nb1; b2 = nb2; b3 = nb3;
    }
    {   // pipeline tail
        const v16bf a = cvt_a_raw(araw);
        acc[0] = wmma_bf16(a, b0, acc[0]);
        acc[1] = wmma_bf16(a, b1, acc[1]);
        acc[2] = wmma_bf16(a, b2, acc[2]);
        acc[3] = wmma_bf16(a, b3, acc[3]);
    }

    const float scale = (z == 0) ? 0.125f : 1.0f;   // hd^-0.5 = 1/8
    const int   h = nsup;                           // wave-uniform head slice
    #pragma unroll
    for (int c4 = 0; c4 < 4; ++c4) {
        const int e  = c4 * 16 + ln;
        const float bn = bias[n0 + e];
        #pragma unroll
        for (int r = 0; r < 8; ++r) {
            const int mr = m0 + r + 8 * hi;
            const int t  = mr >> 3;                 // B = 8
            const int bb = mr & 7;
            const int head = bb * H_NUM + h;
            const unsigned short val = f2bf((acc[c4][r] + bn) * scale);
            if (z == 0)
                Qh[(size_t)head * (T_DIM * HD) + (size_t)t * HD + e] = val;
            else if (z == 1)
                Kh[(size_t)head * (T_DIM * HD) + (size_t)t * HD + e] = val;
            else
                Vt[(size_t)head * (HD * T_DIM) + (size_t)e * T_DIM + t] = val;
        }
    }
}

// ---------------- streaming (flash) attention ----------------
// One wave per (head, 16-query tile): 96 heads * 64 tiles = 6144 waves = 768 blocks.
// V fragments are loaded before the softmax VALU block so they are in flight
// while the online-softmax shuffles/exp execute.
__global__ void __launch_bounds__(256)
attn_kernel(const unsigned short* __restrict__ Qh, const unsigned short* __restrict__ Kh,
            const unsigned short* __restrict__ Vt, unsigned short* __restrict__ Ah)
{
    __shared__ __align__(16) unsigned short plds[8 * 16 * 32];   // 1KB per wave

    const int lane = threadIdx.x & 31;
    const int warp = threadIdx.x >> 5;
    const int wid  = blockIdx.x * 8 + warp;
    const int head = wid >> 6;
    const int t0   = (wid & 63) * 16;
    const int hi   = lane >> 4;
    const int ln   = lane & 15;
    const int ka   = hi ? 8 : 0;
    const int kb   = hi ? 8 : 0;

    const unsigned short* Qp = Qh + (size_t)head * (T_DIM * HD);
    const unsigned short* Kp = Kh + (size_t)head * (T_DIM * HD);
    const unsigned short* Vp = Vt + (size_t)head * (HD * T_DIM);

    // Q tile A-fragments (K-dim = hd = 64 -> two 32-wide k-steps), held in regs
    const unsigned short* qrow = Qp + (size_t)(t0 + ln) * HD;
    const v16bf aq0 = load_a_frag(qrow + ka,      qrow + ka + 16);
    const v16bf aq1 = load_a_frag(qrow + 32 + ka, qrow + 32 + ka + 16);

    float m_run[8], l_run[8];
    #pragma unroll
    for (int r = 0; r < 8; ++r) { m_run[r] = -1e30f; l_run[r] = 0.f; }
    const v8f zero8 = {0.f, 0.f, 0.f, 0.f, 0.f, 0.f, 0.f, 0.f};
    v8f Oacc[4] = {zero8, zero8, zero8, zero8};

    unsigned short* pl = plds + warp * (16 * 32);

    for (int s0 = 0; s0 < T_DIM; s0 += 32) {
        // prefetch next chunk's K rows and V rows into cache (global_prefetch_b8)
        if (s0 + 32 < T_DIM) {
            __builtin_prefetch(Kp + (size_t)(s0 + 32 + lane) * HD, 0, 1);
            __builtin_prefetch(Vp + (size_t)lane        * T_DIM + s0 + 32, 0, 1);
            __builtin_prefetch(Vp + (size_t)(32 + lane) * T_DIM + s0 + 32, 0, 1);
        }

        // ---- K fragments for this 32-key chunk ----
        const unsigned short* kr0 = Kp + (size_t)(s0 + ln)      * HD;
        const unsigned short* kr1 = Kp + (size_t)(s0 + 16 + ln) * HD;
        const v16bf kb00 = load_b_frag(kr0 + kb);
        const v16bf kb01 = load_b_frag(kr0 + 32 + kb);
        const v16bf kb10 = load_b_frag(kr1 + kb);
        const v16bf kb11 = load_b_frag(kr1 + 32 + kb);

        // ---- V fragments issued early: in flight during softmax ----
        v16bf vb[4];
        #pragma unroll
        for (int c4 = 0; c4 < 4; ++c4)
            vb[c4] = load_b_frag(Vp + (size_t)(c4 * 16 + ln) * T_DIM + s0 + kb);

        // ---- scores: two 16x16 tiles ----
        v8f sc0 = zero8, sc1 = zero8;
        sc0 = wmma_bf16(aq0, kb00, sc0);
        sc0 = wmma_bf16(aq1, kb01, sc0);
        sc1 = wmma_bf16(aq0, kb10, sc1);
        sc1 = wmma_bf16(aq1, kb11, sc1);

        // ---- online softmax update (row stats across 16-lane halves) ----
        float pr0[8], pr1[8];
        #pragma unroll
        for (int r = 0; r < 8; ++r) {
            float mx = fmaxf(sc0[r], sc1[r]);
            mx = fmaxf(mx, __shfl_xor(mx, 1));
            mx = fmaxf(mx, __shfl_xor(mx, 2));
            mx = fmaxf(mx, __shfl_xor(mx, 4));
            mx = fmaxf(mx, __shfl_xor(mx, 8));
            const float mnew = fmaxf(m_run[r], mx);
            const float p0 = __expf(sc0[r] - mnew);
            const float p1 = __expf(sc1[r] - mnew);
            float rs = p0 + p1;
            rs += __shfl_xor(rs, 1);
            rs += __shfl_xor(rs, 2);
            rs += __shfl_xor(rs, 4);
            rs += __shfl_xor(rs, 8);
            const float corr = __expf(m_run[r] - mnew);
            l_run[r] = l_run[r] * corr + rs;
            m_run[r] = mnew;
            pr0[r] = p0; pr1[r] = p1;
            Oacc[0][r] *= corr; Oacc[1][r] *= corr;
            Oacc[2][r] *= corr; Oacc[3][r] *= corr;
        }

        // ---- transpose P (C-layout) -> A-fragment via per-wave LDS tile ----
        #pragma unroll
        for (int r = 0; r < 8; ++r) {
            pl[(r + 8 * hi) * 32 + ln]      = f2bf(pr0[r]);
            pl[(r + 8 * hi) * 32 + ln + 16] = f2bf(pr1[r]);
        }
        asm volatile("s_wait_dscnt 0" ::: "memory");
        const v16bf ap = load_a_frag(pl + ln * 32 + ka, pl + ln * 32 + ka + 16);

        // ---- O += P * V ----
        #pragma unroll
        for (int c4 = 0; c4 < 4; ++c4)
            Oacc[c4] = wmma_bf16(ap, vb[c4], Oacc[c4]);
    }

    // ---- epilogue: divide by l, scatter into (T, B, D) bf16 buffer ----
    const int b = head / H_NUM, h = head % H_NUM;
    #pragma unroll
    for (int r = 0; r < 8; ++r) {
        const float inv = 1.0f / l_run[r];
        const int t = t0 + r + 8 * hi;
        #pragma unroll
        for (int c4 = 0; c4 < 4; ++c4) {
            const int e = c4 * 16 + ln;
            Ah[(size_t)(t * B_DIM + b) * D_DIM + h * HD + e] = f2bf(Oacc[c4][r] * inv);
        }
    }
}

// ---------------- output projection: out = Ah @ Wo^T + bo (f32 out) ----------------
// One wave per 16x64 tile, software-pipelined like the input projection.
__global__ void __launch_bounds__(256)
oproj_kernel(const unsigned short* __restrict__ Ah, const unsigned short* __restrict__ Wo,
             const float* __restrict__ bo, float* __restrict__ out)
{
    const int lane = threadIdx.x & 31;
    const int warp = threadIdx.x >> 5;
    const int wid  = blockIdx.x * 8 + warp;     // 0..6143
    const int m0   = (wid / NSUP) * 16;
    const int n0   = (wid % NSUP) * 64;
    const int hi   = lane >> 4;
    const int ln   = lane & 15;
    const int ka   = hi ? 8 : 0;
    const int kb   = hi ? 8 : 0;

    const unsigned short* arow = Ah + (size_t)(m0 + ln) * D_DIM;
    const unsigned short* wr0 = Wo + (size_t)(n0 + ln)      * D_DIM;
    const unsigned short* wr1 = Wo + (size_t)(n0 + 16 + ln) * D_DIM;
    const unsigned short* wr2 = Wo + (size_t)(n0 + 32 + ln) * D_DIM;
    const unsigned short* wr3 = Wo + (size_t)(n0 + 48 + ln) * D_DIM;

    const v8f zero8 = {0.f, 0.f, 0.f, 0.f, 0.f, 0.f, 0.f, 0.f};
    v8f acc[4] = {zero8, zero8, zero8, zero8};

    // ---- software pipeline: stage 0 loads ----
    v16bf a0 = load_a_frag(arow + ka, arow + ka + 16);
    v16bf b0 = load_b_frag(wr0 + kb);
    v16bf b1 = load_b_frag(wr1 + kb);
    v16bf b2 = load_b_frag(wr2 + kb);
    v16bf b3 = load_b_frag(wr3 + kb);

    #pragma unroll 2
    for (int kk = 0; kk < D_DIM - 32; kk += 32) {
        const int kn = kk + 32;
        v16bf na  = load_a_frag(arow + kn + ka, arow + kn + ka + 16);
        v16bf nb0 = load_b_frag(wr0 + kn + kb);
        v16bf nb1 = load_b_frag(wr1 + kn + kb);
        v16bf nb2 = load_b_frag(wr2 + kn + kb);
        v16bf nb3 = load_b_frag(wr3 + kn + kb);

        acc[0] = wmma_bf16(a0, b0, acc[0]);
        acc[1] = wmma_bf16(a0, b1, acc[1]);
        acc[2] = wmma_bf16(a0, b2, acc[2]);
        acc[3] = wmma_bf16(a0, b3, acc[3]);

        a0 = na; b0 = nb0; b1 = nb1; b2 = nb2; b3 = nb3;
    }
    acc[0] = wmma_bf16(a0, b0, acc[0]);
    acc[1] = wmma_bf16(a0, b1, acc[1]);
    acc[2] = wmma_bf16(a0, b2, acc[2]);
    acc[3] = wmma_bf16(a0, b3, acc[3]);

    #pragma unroll
    for (int c4 = 0; c4 < 4; ++c4) {
        const int n  = n0 + c4 * 16 + ln;
        const float bn = bo[n];
        #pragma unroll
        for (int r = 0; r < 8; ++r)
            out[(size_t)(m0 + r + 8 * hi) * D_DIM + n] = acc[c4][r] + bn;
    }
}

// ---------------- host launcher ----------------
extern "C" void kernel_launch(void* const* d_in, const int* in_sizes, int n_in,
                              void* d_out, int out_size, void* d_ws, size_t ws_size,
                              hipStream_t stream) {
    (void)in_sizes; (void)n_in; (void)out_size; (void)ws_size;

    const float* query = (const float*)d_in[0];
    const float* key_  = (const float*)d_in[1];
    const float* value = (const float*)d_in[2];
    const int*   lslen = (const int*)d_in[3];
    const float* Wq  = (const float*)d_in[4];  const float* bq  = (const float*)d_in[5];
    const float* Wk  = (const float*)d_in[6];  const float* bk  = (const float*)d_in[7];
    const float* Wv  = (const float*)d_in[8];  const float* bv  = (const float*)d_in[9];
    const float* Wqm = (const float*)d_in[10]; const float* bqm = (const float*)d_in[11];
    const float* Wkm = (const float*)d_in[12]; const float* bkm = (const float*)d_in[13];
    const float* Wvm = (const float*)d_in[14]; const float* bvm = (const float*)d_in[15];
    const float* Wo  = (const float*)d_in[16]; const float* bo  = (const float*)d_in[17];

    // workspace layout (bf16 / unsigned short elements), total ~58.6 MB
    const size_t WSZ  = (size_t)D_DIM * D_DIM;          // 589824
    const size_t HSZ  = (size_t)NHEAD * T_DIM * HD;     // 6291456
    unsigned short* ws  = (unsigned short*)d_ws;
    unsigned short* Wbf = ws;                 // 7*WSZ : wq,wk,wv,wqm,wkm,wvm,wo
    unsigned short* Qh  = Wbf + 7 * WSZ;      // (head, T, hd)
    unsigned short* Kh  = Qh + HSZ;           // (head, T, hd)
    unsigned short* Vt  = Kh + HSZ;           // (head, hd, T)
    unsigned short* Ah  = Vt + HSZ;           // (T, B, D)

    const float* Wsrc[7] = {Wq, Wk, Wv, Wqm, Wkm, Wvm, Wo};
    for (int i = 0; i < 7; ++i)
        cvt_f32_bf16_kernel<<<(int)((WSZ + 255) / 256), 256, 0, stream>>>(
            Wsrc[i], Wbf + (size_t)i * WSZ, (int)WSZ);

    // 512 M-tiles * 12 N-super-tiles = 6144 wave-tiles -> 768 blocks of 8 waves, z = {q,k,v}
    qkv_proj_kernel<<<dim3(768, 1, 3), 256, 0, stream>>>(
        query, key_, value, Wbf, bq, bk, bv, bqm, bkm, bvm, Qh, Kh, Vt, lslen);

    // 96 heads * 64 query tiles = 6144 waves -> 768 blocks
    attn_kernel<<<768, 256, 0, stream>>>(Qh, Kh, Vt, Ah);

    // final projection into f32 output
    oproj_kernel<<<768, 256, 0, stream>>>(Ah, Wbf + 6 * WSZ, bo, (float*)d_out);
}
